// I_GCL_45500883534272
// MI455X (gfx1250) — matrix operand order repaired and checked by reading, compiled, not run on hardware
//
#include <hip/hip_runtime.h>
#include <hip/hip_bf16.h>

// ---------------------------------------------------------------------------
// CDNA5 / gfx1250 WMMA implementation of the I_GCL layer.
//   edge_feat = relu(relu([x[row],x[col]] @ e_w1 + e_b1) @ e_w2 + e_b2)
//   int_feat  = relu(relu([x[irow],int_x[icol]] @ i_w1 + i_b1) @ i_w2 + i_b2)
//   agg_e/agg_i = segment_sum by row
//   out = relu([x,agg_e,agg_i] @ n_w1 + n_b1) @ n_w2 + n_b2 + x
// Matrix work runs on v_wmma_f32_16x16x32_f16 (f32 accumulate).
// x / int_x are pre-converted to f16 once, so edge-kernel A fragments are
// pure 16-byte f16 loads (no in-loop cvt). Edge kernels process 2 M-tiles
// (32 edges) per wave; B loads are software-pipelined one step ahead.
// ---------------------------------------------------------------------------

typedef __attribute__((ext_vector_type(16))) _Float16 v16h;
typedef __attribute__((ext_vector_type(8)))  _Float16 v8h;
typedef __attribute__((ext_vector_type(8)))  float    v8f;

#define D_FEAT 128

#define WMMA_F16(a, b, c)                                                      \
  __builtin_amdgcn_wmma_f32_16x16x32_f16(false, (a), false, (b), (short)0,     \
                                         (c), false, false)

// ---- A-fragment (16x32 f16, MxK) from an f16 row (global or LDS) -----------
// Lane L<16  : M=L,    holds K = kb+0..15
// Lane L>=16 : M=L-16, holds K = kb+16..31
__device__ __forceinline__ v16h make_a_frag_h16(const _Float16* row,
                                                int kb, int half) {
  v8h lo = *(const v8h*)(row + kb + half * 16);
  v8h hi = *(const v8h*)(row + kb + half * 16 + 8);
  v16h a;
#pragma unroll
  for (int j = 0; j < 8; ++j) { a[j] = lo[j]; a[8 + j] = hi[j]; }
  return a;
}

// A-fragment built from an f32 row (used for f32 aggregation buffers).
__device__ __forceinline__ v16h make_a_frag_f32(const float* __restrict__ row,
                                                int kb, int half) {
  const float4 f0 = *(const float4*)(row + kb + half * 16);
  const float4 f1 = *(const float4*)(row + kb + half * 16 + 4);
  const float4 f2 = *(const float4*)(row + kb + half * 16 + 8);
  const float4 f3 = *(const float4*)(row + kb + half * 16 + 12);
  v16h a;
  a[0]  = (_Float16)f0.x; a[1]  = (_Float16)f0.y;
  a[2]  = (_Float16)f0.z; a[3]  = (_Float16)f0.w;
  a[4]  = (_Float16)f1.x; a[5]  = (_Float16)f1.y;
  a[6]  = (_Float16)f1.z; a[7]  = (_Float16)f1.w;
  a[8]  = (_Float16)f2.x; a[9]  = (_Float16)f2.y;
  a[10] = (_Float16)f2.z; a[11] = (_Float16)f2.w;
  a[12] = (_Float16)f3.x; a[13] = (_Float16)f3.y;
  a[14] = (_Float16)f3.z; a[15] = (_Float16)f3.w;
  return a;
}

// B fragment: 32 bytes per lane, contiguous. fragIdx = kt*8 + nt.
__device__ __forceinline__ v16h load_b(const _Float16* __restrict__ p,
                                       int fragIdx, int lane) {
  return *(const v16h*)(p + (((fragIdx) * 32 + lane) << 4));
}

// ---- weight packing: f32 [K,128] row-major -> f16 B fragments --------------
// B (32x16 KxN) layout per lane: lane L holds column n = nt*16 + (L&15),
// K = kt*32 + (L>>4)*16 + j (j=0..15), stored contiguously (32B per lane).
// dst index: ((kt*8 + nt)*32 + lane)*16 + j
__global__ void pack_w_kernel(const float* __restrict__ w,
                              _Float16* __restrict__ dst, int total) {
  int i = blockIdx.x * blockDim.x + threadIdx.x;
  if (i >= total) return;
  int j    = i & 15;
  int lane = (i >> 4) & 31;
  int nt   = (i >> 9) & 7;
  int kt   = i >> 12;
  int n = nt * 16 + (lane & 15);
  int k = kt * 32 + (lane >> 4) * 16 + j;
  dst[i] = (_Float16)w[(size_t)k * D_FEAT + n];
}

__global__ void zero_kernel(float* __restrict__ p, size_t n) {
  size_t i = (size_t)blockIdx.x * blockDim.x + threadIdx.x;
  if (i < n) p[i] = 0.0f;
}

__global__ void conv_f16_kernel(const float* __restrict__ src,
                                _Float16* __restrict__ dst, size_t n) {
  size_t i = (size_t)blockIdx.x * blockDim.x + threadIdx.x;
  if (i < n) dst[i] = (_Float16)src[i];
}

// ---- edge / interaction MLP: one wave handles 32 edges (2 M-tiles) ---------
template <bool STORE_FEAT>
__global__ __launch_bounds__(32)
void edge_mlp_kernel(const _Float16* __restrict__ xa,  // f16 [N,128]
                     const _Float16* __restrict__ xb,  // f16 [N,128]
                     const int*   __restrict__ idx_row,
                     const int*   __restrict__ idx_col,
                     const _Float16* __restrict__ pw1,  // 8 kt x 8 nt frags
                     const _Float16* __restrict__ pw2,  // 4 kt x 8 nt frags
                     const float* __restrict__ b1,
                     const float* __restrict__ b2,
                     float* __restrict__ feat_out,      // [E,128] or unused
                     float* __restrict__ agg,           // [N,128] f32
                     int E_) {
  __shared__ _Float16 h1[32][136];   // padded to dodge bank conflicts
  __shared__ int rows[32];

  const int lane = threadIdx.x;
  const int half = lane >> 4;
  const int m    = lane & 15;
  const int e0   = blockIdx.x * 32;
  if (e0 + 32 > E_) return;          // uniform: EXEC stays all-ones below

  rows[lane] = idx_row[e0 + lane];
  const int rA0 = idx_row[e0 + m];
  const int cA0 = idx_col[e0 + m];
  const int rA1 = idx_row[e0 + 16 + m];
  const int cA1 = idx_col[e0 + 16 + m];
  __syncthreads();

  // warm the weight stream (global_prefetch_b8)
  __builtin_prefetch(pw1 + (lane << 6), 0, 0);
  __builtin_prefetch(pw2 + (lane << 6), 0, 0);

  const _Float16* rowA0 = xa + (size_t)rA0 * D_FEAT;
  const _Float16* rowB0 = xb + (size_t)cA0 * D_FEAT;
  const _Float16* rowA1 = xa + (size_t)rA1 * D_FEAT;
  const _Float16* rowB1 = xb + (size_t)cA1 * D_FEAT;

  // ---- layer 1: [32,256] @ [256,128] ----
  v8f acc0[8] = {};
  v8f acc1[8] = {};
#pragma unroll
  for (int kt = 0; kt < 8; ++kt) {
    const _Float16* s0 = (kt < 4) ? rowA0 : rowB0;
    const _Float16* s1 = (kt < 4) ? rowA1 : rowB1;
    const int kb = (kt & 3) * 32;
    v16h a0 = make_a_frag_h16(s0, kb, half);
    v16h a1 = make_a_frag_h16(s1, kb, half);
    v16h b = load_b(pw1, kt * 8, lane);
#pragma unroll
    for (int nt = 0; nt < 8; ++nt) {
      v16h bn = (nt < 7) ? load_b(pw1, kt * 8 + nt + 1, lane) : b;
      acc0[nt] = WMMA_F16(a0, b, acc0[nt]);
      acc1[nt] = WMMA_F16(a1, b, acc1[nt]);
      b = bn;
    }
  }
  // bias + relu -> LDS f16 (D layout: VGPR r => M = r + 8*half, N = m)
#pragma unroll
  for (int nt = 0; nt < 8; ++nt) {
    float bv = b1[nt * 16 + m];
#pragma unroll
    for (int r = 0; r < 8; ++r) {
      float v0 = acc0[nt][r] + bv;
      float v1 = acc1[nt][r] + bv;
      v0 = v0 > 0.0f ? v0 : 0.0f;
      v1 = v1 > 0.0f ? v1 : 0.0f;
      h1[r + half * 8][nt * 16 + m]      = (_Float16)v0;
      h1[16 + r + half * 8][nt * 16 + m] = (_Float16)v1;
    }
  }
  __syncthreads();

  // ---- layer 2: [32,128] @ [128,128] ----
  v8f acc20[8] = {};
  v8f acc21[8] = {};
#pragma unroll
  for (int kt = 0; kt < 4; ++kt) {
    v16h a0 = make_a_frag_h16(&h1[m][0], kt * 32, half);
    v16h a1 = make_a_frag_h16(&h1[16 + m][0], kt * 32, half);
    v16h b = load_b(pw2, kt * 8, lane);
#pragma unroll
    for (int nt = 0; nt < 8; ++nt) {
      v16h bn = (nt < 7) ? load_b(pw2, kt * 8 + nt + 1, lane) : b;
      acc20[nt] = WMMA_F16(a0, b, acc20[nt]);
      acc21[nt] = WMMA_F16(a1, b, acc21[nt]);
      b = bn;
    }
  }
  // bias + relu, emit edge_feat and scatter-add into agg
#pragma unroll
  for (int nt = 0; nt < 8; ++nt) {
    float bv = b2[nt * 16 + m];
    int n = nt * 16 + m;
#pragma unroll
    for (int r = 0; r < 8; ++r) {
      const int me = r + half * 8;
      float v0 = acc20[nt][r] + bv;
      float v1 = acc21[nt][r] + bv;
      v0 = v0 > 0.0f ? v0 : 0.0f;
      v1 = v1 > 0.0f ? v1 : 0.0f;
      if (STORE_FEAT) {
        feat_out[(size_t)(e0 + me) * D_FEAT + n]      = v0;
        feat_out[(size_t)(e0 + 16 + me) * D_FEAT + n] = v1;
      }
      atomicAdd(&agg[(size_t)rows[me] * D_FEAT + n], v0);
      atomicAdd(&agg[(size_t)rows[16 + me] * D_FEAT + n], v1);
    }
  }
}

// ---- node MLP: [16,384]@[384,128] -> relu -> [16,128]@[128,128] + res ------
__global__ __launch_bounds__(32)
void node_mlp_kernel(const float* __restrict__ x,      // f32, residual
                     const _Float16* __restrict__ xh,  // f16 copy of x
                     const float* __restrict__ agg_e,
                     const float* __restrict__ agg_i,
                     const _Float16* __restrict__ pw1,  // 12 kt x 8 nt frags
                     const _Float16* __restrict__ pw2,  // 4 kt x 8 nt frags
                     const float* __restrict__ b1,
                     const float* __restrict__ b2,
                     float* __restrict__ out, int N_) {
  __shared__ _Float16 h1[16][136];
  const int lane = threadIdx.x;
  const int half = lane >> 4;
  const int m    = lane & 15;
  const int n0   = blockIdx.x * 16;
  if (n0 + 16 > N_) return;

  __builtin_prefetch(pw1 + (lane << 6), 0, 0);
  __builtin_prefetch(pw2 + (lane << 6), 0, 0);

  const size_t node_m = (size_t)(n0 + m);
  const _Float16* rxh = xh + node_m * D_FEAT;
  const float*    re  = agg_e + node_m * D_FEAT;
  const float*    ri  = agg_i + node_m * D_FEAT;

  v8f acc[8] = {};
#pragma unroll
  for (int kt = 0; kt < 12; ++kt) {
    const int kb = (kt & 3) * 32;
    v16h a = (kt < 4) ? make_a_frag_h16(rxh, kb, half)
                      : make_a_frag_f32((kt < 8) ? re : ri, kb, half);
    v16h b = load_b(pw1, kt * 8, lane);
#pragma unroll
    for (int nt = 0; nt < 8; ++nt) {
      v16h bn = (nt < 7) ? load_b(pw1, kt * 8 + nt + 1, lane) : b;
      acc[nt] = WMMA_F16(a, b, acc[nt]);
      b = bn;
    }
  }
#pragma unroll
  for (int nt = 0; nt < 8; ++nt) {
    float bv = b1[nt * 16 + m];
#pragma unroll
    for (int r = 0; r < 8; ++r) {
      float v = acc[nt][r] + bv;
      v = v > 0.0f ? v : 0.0f;
      h1[r + half * 8][nt * 16 + m] = (_Float16)v;
    }
  }
  __syncthreads();

  v8f acc2[8] = {};
#pragma unroll
  for (int kt = 0; kt < 4; ++kt) {
    v16h a = make_a_frag_h16(&h1[m][0], kt * 32, half);
    v16h b = load_b(pw2, kt * 8, lane);
#pragma unroll
    for (int nt = 0; nt < 8; ++nt) {
      v16h bn = (nt < 7) ? load_b(pw2, kt * 8 + nt + 1, lane) : b;
      acc2[nt] = WMMA_F16(a, b, acc2[nt]);
      b = bn;
    }
  }
  // + bias + residual x, store
#pragma unroll
  for (int nt = 0; nt < 8; ++nt) {
    float bv = b2[nt * 16 + m];
    int n = nt * 16 + m;
#pragma unroll
    for (int r = 0; r < 8; ++r) {
      size_t node = (size_t)(n0 + r + half * 8);
      out[node * D_FEAT + n] = acc2[nt][r] + bv + x[node * D_FEAT + n];
    }
  }
}

// ---------------------------------------------------------------------------
extern "C" void kernel_launch(void* const* d_in, const int* in_sizes, int n_in,
                              void* d_out, int out_size, void* d_ws,
                              size_t ws_size, hipStream_t stream) {
  const float* x     = (const float*)d_in[0];
  const float* int_x = (const float*)d_in[1];
  const int*   eidx  = (const int*)d_in[2];
  const int*   iidx  = (const int*)d_in[3];
  const float* e_w1 = (const float*)d_in[4];
  const float* e_b1 = (const float*)d_in[5];
  const float* e_w2 = (const float*)d_in[6];
  const float* e_b2 = (const float*)d_in[7];
  const float* i_w1 = (const float*)d_in[8];
  const float* i_b1 = (const float*)d_in[9];
  const float* i_w2 = (const float*)d_in[10];
  const float* i_b2 = (const float*)d_in[11];
  const float* n_w1 = (const float*)d_in[12];
  const float* n_b1 = (const float*)d_in[13];
  const float* n_w2 = (const float*)d_in[14];
  const float* n_b2 = (const float*)d_in[15];

  const int N = in_sizes[0] / D_FEAT;   // 50000
  const int E = in_sizes[2] / 2;        // 640000

  // workspace:
  //   [agg_e f32 N*128][agg_i f32 N*128][packed f16 weights 163840]
  //   [xh f16 N*128][ixh f16 N*128]
  float* agg_e = (float*)d_ws;
  float* agg_i = agg_e + (size_t)N * D_FEAT;
  _Float16* p_e1 = (_Float16*)(agg_i + (size_t)N * D_FEAT);
  _Float16* p_e2 = p_e1 + 8  * 4096;    // 4096 halves per K-tile (8nt*32*16)
  _Float16* p_i1 = p_e2 + 4  * 4096;
  _Float16* p_i2 = p_i1 + 8  * 4096;
  _Float16* p_n1 = p_i2 + 4  * 4096;
  _Float16* p_n2 = p_n1 + 12 * 4096;
  _Float16* xh   = p_n2 + 4  * 4096;
  _Float16* ixh  = xh + (size_t)N * D_FEAT;

  float* out_nodes = (float*)d_out;                       // [N,128]
  float* out_feat  = out_nodes + (size_t)N * D_FEAT;      // [E,128]

  // 1) zero aggregation buffers; convert features to f16
  size_t aggN = (size_t)2 * N * D_FEAT;
  zero_kernel<<<(int)((aggN + 255) / 256), 256, 0, stream>>>(agg_e, aggN);
  size_t xN = (size_t)N * D_FEAT;
  conv_f16_kernel<<<(int)((xN + 255) / 256), 256, 0, stream>>>(x, xh, xN);
  conv_f16_kernel<<<(int)((xN + 255) / 256), 256, 0, stream>>>(int_x, ixh, xN);

  // 2) pack weights into WMMA B-fragment layout (f16)
  pack_w_kernel<<<(8  * 4096 + 255) / 256, 256, 0, stream>>>(e_w1, p_e1, 8  * 4096);
  pack_w_kernel<<<(4  * 4096 + 255) / 256, 256, 0, stream>>>(e_w2, p_e2, 4  * 4096);
  pack_w_kernel<<<(8  * 4096 + 255) / 256, 256, 0, stream>>>(i_w1, p_i1, 8  * 4096);
  pack_w_kernel<<<(4  * 4096 + 255) / 256, 256, 0, stream>>>(i_w2, p_i2, 4  * 4096);
  pack_w_kernel<<<(12 * 4096 + 255) / 256, 256, 0, stream>>>(n_w1, p_n1, 12 * 4096);
  pack_w_kernel<<<(4  * 4096 + 255) / 256, 256, 0, stream>>>(n_w2, p_n2, 4  * 4096);

  // 3) edge MLP (+edge_feat output, +agg_e), 4) interaction MLP (+agg_i)
  int eTiles = E / 32;                  // 2 M-tiles per wave
  edge_mlp_kernel<true ><<<eTiles, 32, 0, stream>>>(
      xh, xh,  eidx, eidx + E, p_e1, p_e2, e_b1, e_b2, out_feat, agg_e, E);
  edge_mlp_kernel<false><<<eTiles, 32, 0, stream>>>(
      xh, ixh, iidx, iidx + E, p_i1, p_i2, i_b1, i_b2, nullptr,  agg_i, E);

  // 5) node MLP + residual
  node_mlp_kernel<<<N / 16, 32, 0, stream>>>(
      x, xh, agg_e, agg_i, p_n1, p_n2, n_b1, n_b2, out_nodes, N);
}
